// CNNLSTMProModel_60799557042422
// MI455X (gfx1250) — matrix operand order repaired and checked by reading, not compiled, not run on hardware
//
#include <hip/hip_runtime.h>
#include <hip/hip_bf16.h>

// ---------------------------------------------------------------------------
// Types for CDNA5 WMMA (wave32, 16x16x32 bf16, f32 accum)
// ---------------------------------------------------------------------------
typedef __attribute__((ext_vector_type(16))) __bf16 v16bf;
typedef __attribute__((ext_vector_type(8)))  float  v8f;

#define D_MODEL 512
#define NUM_HEADS 8
#define D_K 64
#define MAX_LEN 500
#define BATCH 32
#define SEQ 480
#define M_ROWS (BATCH * SEQ)      // 15360

__device__ __forceinline__ v8f wmma_bf16(v16bf a, v16bf b, v8f c) {
    return __builtin_amdgcn_wmma_f32_16x16x32_bf16(
        /*neg_a=*/false, a, /*neg_b=*/false, b,
        /*c_mod=*/(short)0, c, /*reuse_a=*/false, /*reuse_b=*/false);
}

union FragU {
    v16bf v;
    uint4 q[2];
};

// A fragment (16 M x 32 K bf16) from a ROW-MAJOR tile (stride ld halves).
// Per CDNA5 layout the per-lane K indices are two contiguous 8-half runs:
//   j=0..7  -> k = g*8 + j        j=8..15 -> k = 16 + g*8 + (j-8)
// => exactly two 16B LDS loads per lane.
__device__ __forceinline__ v16bf load_a_rowmajor(const __bf16* tile, int ld) {
    const int lane = threadIdx.x & 31;
    const int m = lane & 15, g = lane >> 4;
    FragU u;
    u.q[0] = *(const uint4*)(tile + m * ld + g * 8);
    u.q[1] = *(const uint4*)(tile + m * ld + 16 + g * 8);
    return u.v;
}

// B fragment (32 K x 16 N bf16) from a K-MAJOR tile, i.e. tile[n*ld + k]:
//   j=0..7 -> k = g*8 + j, j=8..15 -> k = 16 + g*8 + (j-8), n = lane&15.
__device__ __forceinline__ v16bf load_b_kmajor(const __bf16* tile, int ld) {
    const int lane = threadIdx.x & 31;
    const int n = lane & 15, g = lane >> 4;
    FragU u;
    u.q[0] = *(const uint4*)(tile + n * ld + g * 8);
    u.q[1] = *(const uint4*)(tile + n * ld + 16 + g * 8);
    return u.v;
}

__device__ __forceinline__ float rmax16(float v) {
#pragma unroll
    for (int m = 1; m < 16; m <<= 1) v = fmaxf(v, __shfl_xor(v, m, 32));
    return v;
}
__device__ __forceinline__ float rsum16(float v) {
#pragma unroll
    for (int m = 1; m < 16; m <<= 1) v += __shfl_xor(v, m, 32);
    return v;
}

// ---------------------------------------------------------------------------
// Converters
// ---------------------------------------------------------------------------
__global__ void cvt_f32_bf16(const float* __restrict__ in,
                             __bf16* __restrict__ out, size_t n) {
    size_t i = (size_t)blockIdx.x * blockDim.x + threadIdx.x;
    const size_t stride = (size_t)gridDim.x * blockDim.x;
    for (; i < n; i += stride) out[i] = (__bf16)in[i];
}

// 512x512 weight: convert + transpose -> out[n*512 + k] = in[k*512 + n]
__global__ void cvt_f32_bf16_T(const float* __restrict__ in,
                               __bf16* __restrict__ out) {
    size_t i = (size_t)blockIdx.x * blockDim.x + threadIdx.x;
    const size_t stride = (size_t)gridDim.x * blockDim.x;
    const size_t n = (size_t)D_MODEL * D_MODEL;
    for (; i < n; i += stride) {
        const size_t k = i >> 9, c = i & 511;
        out[c * D_MODEL + k] = (__bf16)in[i];
    }
}

__global__ void zero_f32(float* __restrict__ p, size_t n) {
    size_t i = (size_t)blockIdx.x * blockDim.x + threadIdx.x;
    const size_t stride = (size_t)gridDim.x * blockDim.x;
    for (; i < n; i += stride) p[i] = 0.0f;
}

// ---------------------------------------------------------------------------
// GEMM: Y = X(bf16)[15360x512] @ W + b, with W given TRANSPOSED (Wt[n][k]).
// WG = 128 threads (4 waves); tile 64(M) x 64(N); K step 32.
// vt == 0 : head-split output [B,H,S,64] bf16   (Q, K)
// vt == 1 : head-split + transposed [B,H,64,S] bf16  (V, for ctx-GEMM B-frags)
// ---------------------------------------------------------------------------
__global__ __launch_bounds__(128)
void gemm_qkv(const __bf16* __restrict__ X, const __bf16* __restrict__ Wt,
              const float* __restrict__ bias, __bf16* __restrict__ Out,
              const int vt) {
    __shared__ __attribute__((aligned(16))) __bf16 sA[64 * 32];   // [m][k]
    __shared__ __attribute__((aligned(16))) __bf16 sBt[64 * 32];  // [n][k]

    const int tid  = threadIdx.x;
    const int wave = tid >> 5;
    const int lane = tid & 31;
    const int grp  = lane >> 4;
    const int hn   = lane & 15;
    const int row0 = blockIdx.x * 64;
    const int col0 = blockIdx.y * 64;

    v8f acc[4] = {};

    for (int k0 = 0; k0 < D_MODEL; k0 += 32) {
        __syncthreads();
#pragma unroll
        for (int i = tid; i < 256; i += 128) {
            const int r = i >> 2, c = i & 3;
            ((uint4*)sA)[i] =
                ((const uint4*)(X + (size_t)(row0 + r) * D_MODEL + k0))[c];
            ((uint4*)sBt)[i] =
                ((const uint4*)(Wt + (size_t)(col0 + r) * D_MODEL + k0))[c];
        }
        __syncthreads();

        const v16bf a = load_a_rowmajor(&sA[wave * 16 * 32], 32);
#pragma unroll
        for (int nt = 0; nt < 4; ++nt) {
            const v16bf b = load_b_kmajor(&sBt[nt * 16 * 32], 32);
            acc[nt] = wmma_bf16(a, b, acc[nt]);
        }
    }

#pragma unroll
    for (int nt = 0; nt < 4; ++nt) {
#pragma unroll
        for (int r = 0; r < 8; ++r) {
            const int m = row0 + wave * 16 + r + 8 * grp;   // flat row b*S+s
            const int n = col0 + nt * 16 + hn;              // model dim
            const float v = acc[nt][r] + bias[n];
            const int b = m / SEQ, s = m % SEQ;
            const int h = n >> 6, d = n & 63;
            const size_t hb = (size_t)b * NUM_HEADS + h;
            if (vt == 0)
                Out[(hb * SEQ + s) * D_K + d] = (__bf16)v;
            else
                Out[(hb * D_K + d) * SEQ + s] = (__bf16)v;
        }
    }
}

// ---------------------------------------------------------------------------
// Output projection GEMM (f32 out + bias, flat [15360, 512]); Wt transposed.
// ---------------------------------------------------------------------------
__global__ __launch_bounds__(128)
void gemm_out(const __bf16* __restrict__ X, const __bf16* __restrict__ Wt,
              const float* __restrict__ bias, float* __restrict__ Out) {
    __shared__ __attribute__((aligned(16))) __bf16 sA[64 * 32];
    __shared__ __attribute__((aligned(16))) __bf16 sBt[64 * 32];

    const int tid  = threadIdx.x;
    const int wave = tid >> 5;
    const int lane = tid & 31;
    const int grp  = lane >> 4;
    const int hn   = lane & 15;
    const int row0 = blockIdx.x * 64;
    const int col0 = blockIdx.y * 64;

    v8f acc[4] = {};

    for (int k0 = 0; k0 < D_MODEL; k0 += 32) {
        __syncthreads();
#pragma unroll
        for (int i = tid; i < 256; i += 128) {
            const int r = i >> 2, c = i & 3;
            ((uint4*)sA)[i] =
                ((const uint4*)(X + (size_t)(row0 + r) * D_MODEL + k0))[c];
            ((uint4*)sBt)[i] =
                ((const uint4*)(Wt + (size_t)(col0 + r) * D_MODEL + k0))[c];
        }
        __syncthreads();

        const v16bf a = load_a_rowmajor(&sA[wave * 16 * 32], 32);
#pragma unroll
        for (int nt = 0; nt < 4; ++nt) {
            const v16bf b = load_b_kmajor(&sBt[nt * 16 * 32], 32);
            acc[nt] = wmma_bf16(a, b, acc[nt]);
        }
    }

#pragma unroll
    for (int nt = 0; nt < 4; ++nt) {
#pragma unroll
        for (int r = 0; r < 8; ++r) {
            const int m = row0 + wave * 16 + r + 8 * grp;
            const int n = col0 + nt * 16 + hn;
            Out[(size_t)m * D_MODEL + n] = acc[nt][r] + bias[n];
        }
    }
}

// ---------------------------------------------------------------------------
// Causal attention with relative bias.
// WG = 96 threads (3 waves); owns (batch b, 48 query rows); loops all 8 heads
// sequentially (race-free attn-mean accumulation). Two-pass online softmax.
// K staged row-major [key][dk] (k-major for Q*K^T B-frags); V staged from the
// pre-transposed [B,H,dk,S] tensor as [dk][key] (k-major for P*V B-frags).
// ---------------------------------------------------------------------------
#define QB 48
#define ATTN_THREADS 96

__global__ __launch_bounds__(ATTN_THREADS)
void attn_kernel(const __bf16* __restrict__ Qb, const __bf16* __restrict__ Kb,
                 const __bf16* __restrict__ Vt,
                 const float* __restrict__ rel_bias,
                 __bf16* __restrict__ ctxb, float* __restrict__ attn_mean) {
    __shared__ __attribute__((aligned(16))) __bf16 sQ[QB * D_K];   // [m][dk]
    __shared__ __attribute__((aligned(16))) __bf16 sK[32 * D_K];   // [key][dk]
    __shared__ __attribute__((aligned(16))) __bf16 sVt[D_K * 32];  // [dk][key]
    __shared__ __attribute__((aligned(16))) __bf16 sP[3][16 * 32]; // [m][key]

    const int wg   = blockIdx.x;              // BATCH * (SEQ/QB)
    const int b    = wg / (SEQ / QB);
    const int q0   = (wg % (SEQ / QB)) * QB;
    const int tid  = threadIdx.x;
    const int wave = tid >> 5;
    const int lane = tid & 31;
    const int grp  = lane >> 4;
    const int hn   = lane & 15;
    const int qw   = q0 + wave * 16;

    const int nkt = (q0 + QB + 31) / 32;      // 32-key blocks (uniform per WG)
    const float scale = 0.125f;               // 1/sqrt(64)
    const float hinv  = 1.0f / (float)NUM_HEADS;

    for (int h = 0; h < NUM_HEADS; ++h) {
        const size_t hb = (size_t)b * NUM_HEADS + h;
        const __bf16* Qh  = Qb + hb * SEQ * D_K;   // [s][dk]
        const __bf16* Kh  = Kb + hb * SEQ * D_K;   // [s][dk]
        const __bf16* Vth = Vt + hb * D_K * SEQ;   // [dk][s]

        // ---- stage Q rows (48 x 64 bf16) ----
        __syncthreads();
#pragma unroll
        for (int i = tid; i < QB * 8; i += ATTN_THREADS) {
            const int r = i >> 3, c = i & 7;
            ((uint4*)sQ)[i] = ((const uint4*)(Qh + (size_t)(q0 + r) * D_K))[c];
        }
        __syncthreads();

        const v16bf aq0 = load_a_rowmajor(&sQ[wave * 16 * D_K], D_K);      // k 0..31
        const v16bf aq1 = load_a_rowmajor(&sQ[wave * 16 * D_K + 32], D_K); // k 32..63

        // ---- pass 1: online row max + exp-sum ----
        float mrow[8], lrow[8];
#pragma unroll
        for (int r = 0; r < 8; ++r) { mrow[r] = -3.0e38f; lrow[r] = 0.0f; }

        for (int kt = 0; kt < nkt; ++kt) {
            const int kb0 = kt * 32;
            __syncthreads();
#pragma unroll
            for (int i = tid; i < 256; i += ATTN_THREADS)
                ((uint4*)sK)[i] = ((const uint4*)(Kh + (size_t)kb0 * D_K))[i];
            __builtin_prefetch(Kh + (size_t)(kb0 + 32) * D_K, 0, 0);
            __syncthreads();

#pragma unroll
            for (int t = 0; t < 2; ++t) {
                v8f s = {};
                s = wmma_bf16(aq0, load_b_kmajor(&sK[t * 16 * D_K], D_K), s);
                s = wmma_bf16(aq1, load_b_kmajor(&sK[t * 16 * D_K + 32], D_K), s);
#pragma unroll
                for (int r = 0; r < 8; ++r) {
                    const int m   = r + 8 * grp;
                    const int q   = qw + m;
                    const int key = kb0 + t * 16 + hn;
                    float val = s[r] * scale +
                                rel_bias[((size_t)h * MAX_LEN + q) * MAX_LEN + key];
                    val = (key > q) ? -1.0e30f : val;
                    const float tmax = rmax16(val);
                    const float mold = mrow[r];
                    const float mnew = fmaxf(mold, tmax);
                    const float ps   = rsum16(__expf(val - mnew));
                    lrow[r] = lrow[r] * __expf(mold - mnew) + ps;
                    mrow[r] = mnew;
                }
            }
        }

        float invl[8];
#pragma unroll
        for (int r = 0; r < 8; ++r) invl[r] = 1.0f / lrow[r];

        // ---- pass 2: probabilities -> attn mean + context GEMM ----
        v8f cacc[4] = {};
        for (int kt = 0; kt < nkt; ++kt) {
            const int kb0 = kt * 32;
            __syncthreads();
#pragma unroll
            for (int i = tid; i < 256; i += ATTN_THREADS) {
                ((uint4*)sK)[i] = ((const uint4*)(Kh + (size_t)kb0 * D_K))[i];
                // sVt: row d (i>>2) of Vt, keys kb0 + (i&3)*8 .. +8
                const int d = i >> 2, c = i & 3;
                ((uint4*)sVt)[i] =
                    *(const uint4*)(Vth + (size_t)d * SEQ + kb0 + c * 8);
            }
            __syncthreads();

#pragma unroll
            for (int t = 0; t < 2; ++t) {
                v8f s = {};
                s = wmma_bf16(aq0, load_b_kmajor(&sK[t * 16 * D_K], D_K), s);
                s = wmma_bf16(aq1, load_b_kmajor(&sK[t * 16 * D_K + 32], D_K), s);
#pragma unroll
                for (int r = 0; r < 8; ++r) {
                    const int m   = r + 8 * grp;
                    const int q   = qw + m;
                    const int key = kb0 + t * 16 + hn;
                    float val = s[r] * scale +
                                rel_bias[((size_t)h * MAX_LEN + q) * MAX_LEN + key];
                    val = (key > q) ? -1.0e30f : val;
                    const float p = __expf(val - mrow[r]) * invl[r];
                    float* am = attn_mean + ((size_t)(b * SEQ + q)) * SEQ + key;
                    *am = *am + p * hinv;   // race-free: WG owns these rows
                    sP[wave][m * 32 + t * 16 + hn] = (__bf16)p;
                }
            }
            __syncthreads();  // P visible across lanes

            const v16bf ap = load_a_rowmajor(&sP[wave][0], 32);
#pragma unroll
            for (int nt = 0; nt < 4; ++nt) {
                const v16bf bv = load_b_kmajor(&sVt[nt * 16 * 32], 32);
                cacc[nt] = wmma_bf16(ap, bv, cacc[nt]);
            }
        }

        // ---- store context for this head: ctx[b][q][h*64 + d] (bf16) ----
#pragma unroll
        for (int nt = 0; nt < 4; ++nt) {
#pragma unroll
            for (int r = 0; r < 8; ++r) {
                const int q = qw + r + 8 * grp;
                const int d = nt * 16 + hn;
                ctxb[((size_t)(b * SEQ + q)) * D_MODEL + h * D_K + d] =
                    (__bf16)cacc[nt][r];
            }
        }
    }
}

// ---------------------------------------------------------------------------
// Host launcher
// ---------------------------------------------------------------------------
extern "C" void kernel_launch(void* const* d_in, const int* in_sizes, int n_in,
                              void* d_out, int out_size, void* d_ws, size_t ws_size,
                              hipStream_t stream) {
    const float* x        = (const float*)d_in[0];
    const float* wq       = (const float*)d_in[1];
    const float* bq       = (const float*)d_in[2];
    const float* wk       = (const float*)d_in[3];
    const float* bk       = (const float*)d_in[4];
    const float* wv       = (const float*)d_in[5];
    const float* bv       = (const float*)d_in[6];
    const float* wo       = (const float*)d_in[7];
    const float* bo       = (const float*)d_in[8];
    const float* rel_bias = (const float*)d_in[9];

    const size_t NX = (size_t)M_ROWS * D_MODEL;   // 7,864,320
    const size_t NW = (size_t)D_MODEL * D_MODEL;  // 262,144

    __bf16* xb   = (__bf16*)d_ws;
    __bf16* wqt  = xb  + NX;
    __bf16* wkt  = wqt + NW;
    __bf16* wvt  = wkt + NW;
    __bf16* wot  = wvt + NW;
    __bf16* Qb   = wot + NW;     // [B,H,S,64]
    __bf16* Kb   = Qb  + NX;     // [B,H,S,64]
    __bf16* Vt   = Kb  + NX;     // [B,H,64,S]  (transposed)
    __bf16* ctxb = Vt  + NX;     // [B*S, 512]

    float* out_proj  = (float*)d_out;             // [B, S, 512]
    float* attn_part = out_proj + NX;             // [B, S, S]
    const size_t attn_n = (size_t)BATCH * SEQ * SEQ;

    // 1) convert x; convert+transpose weights
    cvt_f32_bf16<<<2048, 256, 0, stream>>>(x, xb, NX);
    cvt_f32_bf16_T<<<512, 256, 0, stream>>>(wq, wqt);
    cvt_f32_bf16_T<<<512, 256, 0, stream>>>(wk, wkt);
    cvt_f32_bf16_T<<<512, 256, 0, stream>>>(wv, wvt);
    cvt_f32_bf16_T<<<512, 256, 0, stream>>>(wo, wot);

    // 2) QKV projections
    dim3 ggrid(M_ROWS / 64, D_MODEL / 64);        // 240 x 8
    gemm_qkv<<<ggrid, 128, 0, stream>>>(xb, wqt, bq, Qb, 0);
    gemm_qkv<<<ggrid, 128, 0, stream>>>(xb, wkt, bk, Kb, 0);
    gemm_qkv<<<ggrid, 128, 0, stream>>>(xb, wvt, bv, Vt, 1);  // V transposed

    // 3) zero attn-mean region (causal upper triangle stays exactly 0)
    zero_f32<<<2048, 256, 0, stream>>>(attn_part, attn_n);

    // 4) attention
    attn_kernel<<<BATCH * (SEQ / QB), ATTN_THREADS, 0, stream>>>(
        Qb, Kb, Vt, rel_bias, ctxb, attn_part);

    // 5) output projection -> f32
    gemm_out<<<ggrid, 128, 0, stream>>>(ctxb, wot, bo, out_proj);
}
